// HybridGNN_70497593197185
// MI455X (gfx1250) — matrix operand order repaired and checked by reading, not compiled, not run on hardware
//
#include <hip/hip_runtime.h>
#include <math.h>

// ---------------------------------------------------------------------------
// HybridGNN forward for MI455X (gfx1250, wave32, WMMA).
// GEMMs (≈170 GFLOP total) run on v_wmma_f32_16x16x32_f16 with f32 accum.
// Graph ops (segment mean, GraphNorm, GAT edge softmax) are VALU/atomic bound.
// Input pointer mapping assumes depth-first insertion order of setup_inputs():
//   0:x 1:edge_index 2:batch, then params: sage1..5{Wl,bl,Wr}, snorm1..4{w,b,ms},
//   gat1..4{W,a_src,a_dst,bias}, gnorm1..3{w,b,ms}, cls{W,b}   (57 inputs)
// ---------------------------------------------------------------------------

#define NN    8192
#define NE    65536
#define NG    4
#define INF_  50
#define HID_  512
#define OUTF  121
#define HEADS_ 4
#define EPS_  1e-5f

typedef __attribute__((ext_vector_type(16))) _Float16 v16h;
typedef __attribute__((ext_vector_type(8)))  _Float16 v8h;
typedef __attribute__((ext_vector_type(8)))  float    v8f;

__device__ __forceinline__ v8f vzero8() {
  v8f z;
#pragma unroll
  for (int i = 0; i < 8; ++i) z[i] = 0.f;
  return z;
}

__device__ __forceinline__ void atomicMaxF(float* a, float v) {
  // sign-aware bit trick: positives via signed max, negatives via unsigned min
  if (v >= 0.f) atomicMax((int*)a, __float_as_int(v));
  else          atomicMin((unsigned int*)a, __float_as_uint(v));
}

// ---------------------------------------------------------------------------
// WMMA GEMM: C[M=8192, nvalid] = A[M, K] * W^T (+bias) (+accumulate)
// A: f16 row-major, lda = Kpad (multiple of 32). Bpk: packed B fragments.
// Packed B layout per 16(N)x32(K) tile: lane l <-> n offset l%16 (l/16 picks k half),
// element e (0..15) <-> k offset; 512 f16 per tile, tiles ordered [ntile][ktile].
// Block: 256 thr = 8 waves; block tile 128x64; wave tile 32x32 (2x2 frags).
// flags: bit0 = accumulate into C, bit1 = add bias[col].
// ---------------------------------------------------------------------------
__global__ __launch_bounds__(256) void k_gemm(
    const _Float16* __restrict__ A, int lda,
    const _Float16* __restrict__ Bpk, int ktiles,
    const float* __restrict__ bias,
    float* __restrict__ C, int ldc,
    int nvalid, int flags)
{
  const int lane = threadIdx.x & 31;
  const int wave = threadIdx.x >> 5;
  const int wm = wave >> 1;            // 0..3 along M
  const int wn = wave & 1;             // 0..1 along N
  const int row0 = blockIdx.x * 128 + wm * 32;
  const int col0 = blockIdx.y * 64 + wn * 32;

  const int r  = lane & 15;            // row within 16-tile (A) / col (C)
  const int hi = lane >> 4;            // half-wave selector

  const _Float16* a0p = A + (size_t)(row0 + r) * lda;
  const _Float16* a1p = a0p + (size_t)16 * lda;

  const int nt0 = col0 >> 4;
  const _Float16* b0p = Bpk + ((size_t)nt0 * ktiles) * 512 + lane * 16;
  const _Float16* b1p = Bpk + ((size_t)(nt0 + 1) * ktiles) * 512 + lane * 16;

  v8f c00 = vzero8(), c01 = vzero8(), c10 = vzero8(), c11 = vzero8();

  for (int kt = 0; kt < ktiles; ++kt) {
    const int k0 = kt * 32;
    // 16-bit A fragment (16x32 MxK): lane r holds row; VGPR0-3 = k {hi*8..hi*8+7},
    // VGPR4-7 = k {16+hi*8..}. Two 16B contiguous loads per fragment per lane.
    union { v16h v; v8h h[2]; } a0, a1, b0, b1;
    a0.h[0] = *(const v8h*)(a0p + k0 + hi * 8);
    a0.h[1] = *(const v8h*)(a0p + k0 + 16 + hi * 8);
    a1.h[0] = *(const v8h*)(a1p + k0 + hi * 8);
    a1.h[1] = *(const v8h*)(a1p + k0 + 16 + hi * 8);
    b0.v = *(const v16h*)(b0p + (size_t)kt * 512);
    b1.v = *(const v16h*)(b1p + (size_t)kt * 512);

    if (kt + 2 < ktiles) {                    // feed global_prefetch_b8
      __builtin_prefetch(a0p + k0 + 64, 0, 0);
      __builtin_prefetch(b0p + (size_t)(kt + 2) * 512, 0, 0);
      __builtin_prefetch(b1p + (size_t)(kt + 2) * 512, 0, 0);
    }

    c00 = __builtin_amdgcn_wmma_f32_16x16x32_f16(false, a0.v, false, b0.v, (short)0, c00, false, false);
    c01 = __builtin_amdgcn_wmma_f32_16x16x32_f16(false, a0.v, false, b1.v, (short)0, c01, false, false);
    c10 = __builtin_amdgcn_wmma_f32_16x16x32_f16(false, a1.v, false, b0.v, (short)0, c10, false, false);
    c11 = __builtin_amdgcn_wmma_f32_16x16x32_f16(false, a1.v, false, b1.v, (short)0, c11, false, false);
  }

  // C/D layout: VGPR i, lanes 0-15 -> M=i, N=lane; lanes 16-31 -> M=8+i.
  const int rofs = hi * 8;
  v8f* accs[4] = { &c00, &c01, &c10, &c11 };
#pragma unroll
  for (int t = 0; t < 4; ++t) {
    const int tm = t >> 1, tn = t & 1;
    const int col = col0 + tn * 16 + r;
    if (col < nvalid) {
      const float bv = (flags & 2) ? bias[col] : 0.f;
      const int rb = row0 + tm * 16 + rofs;
#pragma unroll
      for (int i = 0; i < 8; ++i) {
        float v = (*accs[t])[i] + bv;
        float* cp = C + (size_t)(rb + i) * ldc + col;
        if (flags & 1) *cp += v; else *cp = v;
      }
    }
  }
}

// ---------------------------------------------------------------------------
// Helper kernels (VALU / atomics)
// ---------------------------------------------------------------------------
__global__ void k_fill_f32(float* p, long long n, float v) {
  long long i = (long long)blockIdx.x * blockDim.x + threadIdx.x;
  if (i < n) p[i] = v;
}

__global__ void k_graph_count(const int* __restrict__ batch, int n, int* cnt) {
  int i = blockIdx.x * blockDim.x + threadIdx.x;
  if (i < n) atomicAdd(&cnt[batch[i]], 1);
}

__global__ void k_prefix(const int* __restrict__ cnt, int* gs) {
  if (threadIdx.x == 0 && blockIdx.x == 0) {
    gs[0] = 0;
    for (int g = 0; g < NG; ++g) gs[g + 1] = gs[g] + cnt[g];
  }
}

__global__ void k_deg(const int* __restrict__ dst, int e, int* deg) {
  int i = blockIdx.x * blockDim.x + threadIdx.x;
  if (i < e) atomicAdd(&deg[dst[i]], 1);
}

__global__ void k_scatter_add(const float* __restrict__ X, int ld,
                              const int* __restrict__ src, const int* __restrict__ dst,
                              int e, int f, float* __restrict__ AGG) {
  long long i = (long long)blockIdx.x * blockDim.x + threadIdx.x;
  long long tot = (long long)e * f;
  if (i >= tot) return;
  int ei = (int)(i / f), fi = (int)(i % f);
  atomicAdd(&AGG[(size_t)dst[ei] * ld + fi], X[(size_t)src[ei] * ld + fi]);
}

__global__ void k_div_deg(float* __restrict__ AGG, int f, const int* __restrict__ deg, int n) {
  long long i = (long long)blockIdx.x * blockDim.x + threadIdx.x;
  long long tot = (long long)n * f;
  if (i >= tot) return;
  int r = (int)(i / f);
  AGG[i] /= fmaxf((float)deg[r], 1.f);
}

__global__ void k_cvt_f16(const float* __restrict__ in, int ldin, int cols,
                          _Float16* __restrict__ out, int ldout, int rows) {
  long long i = (long long)blockIdx.x * blockDim.x + threadIdx.x;
  long long tot = (long long)rows * ldout;
  if (i >= tot) return;
  int r = (int)(i / ldout), c = (int)(i % ldout);
  float v = (c < cols) ? in[(size_t)r * ldin + c] : 0.f;
  out[i] = (_Float16)v;
}

__global__ void k_concat_cvt(const float* __restrict__ s, const float* __restrict__ g,
                             _Float16* __restrict__ out) {
  long long i = (long long)blockIdx.x * blockDim.x + threadIdx.x;
  long long tot = (long long)NN * 256;
  if (i >= tot) return;
  int r = (int)(i / 256), c = (int)(i % 256);
  float v = 0.f;
  if (c < OUTF) v = s[(size_t)r * OUTF + c];
  else if (c < 2 * OUTF) v = g[(size_t)r * OUTF + (c - OUTF)];
  out[i] = (_Float16)v;
}

// pack W (nout x korig, row major, f32) -> B fragments (lane<->n, elem<->k), zero-padded
__global__ void k_pack_w(const float* __restrict__ W, int nout, int korig,
                         int ktiles, int ntiles, _Float16* __restrict__ out) {
  int idx = blockIdx.x * blockDim.x + threadIdx.x;
  int tot = ntiles * ktiles * 32;
  if (idx >= tot) return;
  int lane = idx & 31;
  int t = idx >> 5;
  int kt = t % ktiles, nt = t / ktiles;
  int n = nt * 16 + (lane & 15);
  _Float16* o = out + (size_t)idx * 16;
#pragma unroll
  for (int e2 = 0; e2 < 16; ++e2) {
    int k = kt * 32 + ((e2 < 8) ? ((lane >> 4) * 8 + e2) : (16 + (lane >> 4) * 8 + (e2 - 8)));
    float v = (n < nout && k < korig) ? W[(size_t)n * korig + k] : 0.f;
    o[e2] = (_Float16)v;
  }
}

// per-(graph, feature) mean or mean-of-squares over sorted contiguous node range
__global__ void k_seg_stat(const float* __restrict__ X, int ld, int F,
                           const int* __restrict__ gs, float* __restrict__ out, int sq) {
  int i = blockIdx.x * blockDim.x + threadIdx.x;
  if (i >= NG * F) return;
  int g = i / F, f = i % F;
  int s0 = gs[g], s1 = gs[g + 1];
  float s = 0.f;
  for (int n = s0; n < s1; ++n) {
    float v = X[(size_t)n * ld + f];
    s += sq ? v * v : v;
  }
  out[i] = s / fmaxf((float)(s1 - s0), 1.f);
}

__global__ void k_center(float* __restrict__ X, int F, const int* __restrict__ batch,
                         const float* __restrict__ ms, const float* __restrict__ mean, int n) {
  long long i = (long long)blockIdx.x * blockDim.x + threadIdx.x;
  long long tot = (long long)n * F;
  if (i >= tot) return;
  int r = (int)(i / F), f = (int)(i % F);
  X[i] -= ms[f] * mean[batch[r] * F + f];
}

__global__ void k_norm_act(float* __restrict__ X, int F, const int* __restrict__ batch,
                           const float* __restrict__ w, const float* __restrict__ b,
                           const float* __restrict__ var, int act, int n) {
  long long i = (long long)blockIdx.x * blockDim.x + threadIdx.x;
  long long tot = (long long)n * F;
  if (i >= tot) return;
  int r = (int)(i / F), f = (int)(i % F);
  float v = w[f] * X[i] * rsqrtf(var[batch[r] * F + f] + EPS_) + b[f];
  if (act == 1) v = fmaxf(v, 0.f);
  else if (act == 2) v = (v > 0.f) ? v : (__expf(v) - 1.f);
  X[i] = v;
}

// per-(node, head) attention dots, one wave32 per (node, head)
__global__ void k_attdot(const float* __restrict__ HB, int HC, int C, int H,
                         const float* __restrict__ asrc, const float* __restrict__ adst,
                         float* __restrict__ as_, float* __restrict__ ad_, int n) {
  int gt = blockIdx.x * blockDim.x + threadIdx.x;
  int warp = gt >> 5;
  int lane = gt & 31;
  if (warp >= n * H) return;
  int node = warp / H, h = warp % H;
  const float* hp = HB + (size_t)node * HC + h * C;
  float s1 = 0.f, s2 = 0.f;
  for (int c = lane; c < C; c += 32) {
    float hv = hp[c];
    s1 += hv * asrc[h * C + c];
    s2 += hv * adst[h * C + c];
  }
  for (int m = 16; m > 0; m >>= 1) {
    s1 += __shfl_xor(s1, m, 32);
    s2 += __shfl_xor(s2, m, 32);
  }
  if (lane == 0) { as_[node * H + h] = s1; ad_[node * H + h] = s2; }
}

// leaky-relu edge scores (+ self loops) and running segment max
__global__ void k_score(const int* __restrict__ src, const int* __restrict__ dst,
                        int e, int n, int H,
                        const float* __restrict__ as_, const float* __restrict__ ad_,
                        float* __restrict__ esc, float* __restrict__ mx) {
  int i = blockIdx.x * blockDim.x + threadIdx.x;
  if (i >= e + n) return;
  int s_ = (i < e) ? src[i] : (i - e);
  int d_ = (i < e) ? dst[i] : (i - e);
  for (int h = 0; h < H; ++h) {
    float v = as_[s_ * H + h] + ad_[d_ * H + h];
    v = (v > 0.f) ? v : 0.2f * v;
    esc[(size_t)i * H + h] = v;
    atomicMaxF(&mx[d_ * H + h], v);
  }
}

__global__ void k_expw(const int* __restrict__ src, const int* __restrict__ dst,
                       int e, int n, int H,
                       float* __restrict__ esc, const float* __restrict__ mx,
                       float* __restrict__ den) {
  int i = blockIdx.x * blockDim.x + threadIdx.x;
  if (i >= e + n) return;
  int d_ = (i < e) ? dst[i] : (i - e);
  for (int h = 0; h < H; ++h) {
    float w = __expf(esc[(size_t)i * H + h] - mx[d_ * H + h]);
    esc[(size_t)i * H + h] = w;
    atomicAdd(&den[d_ * H + h], w);
  }
}

__global__ void k_gat_agg(const int* __restrict__ src, const int* __restrict__ dst,
                          int e, int n, int H, int C,
                          const float* __restrict__ HB, const float* __restrict__ esc,
                          const float* __restrict__ den, float* __restrict__ OUT) {
  long long i = (long long)blockIdx.x * blockDim.x + threadIdx.x;
  const int HC = H * C;
  long long tot = (long long)(e + n) * HC;
  if (i >= tot) return;
  int ed = (int)(i / HC);
  int j = (int)(i % HC);
  int h = j / C;
  int s_ = (ed < e) ? src[ed] : (ed - e);
  int d_ = (ed < e) ? dst[ed] : (ed - e);
  float a = esc[(size_t)ed * H + h] / (den[d_ * H + h] + 1e-16f);
  atomicAdd(&OUT[(size_t)d_ * HC + j], a * HB[(size_t)s_ * HC + j]);
}

__global__ void k_add_bias(float* __restrict__ X, int F, const float* __restrict__ b, int n) {
  long long i = (long long)blockIdx.x * blockDim.x + threadIdx.x;
  long long tot = (long long)n * F;
  if (i >= tot) return;
  X[i] += b[i % F];
}

// ---------------------------------------------------------------------------
// Host orchestration
// ---------------------------------------------------------------------------
static inline unsigned gblk(long long t) { return (unsigned)((t + 255) / 256); }

static void run_gemm(hipStream_t st, const _Float16* A, int Kpad, const _Float16* Wpk,
                     const float* bias, float* C, int ldc, int nvalid, int Npad, int flags) {
  dim3 grid(NN / 128, Npad / 64);
  k_gemm<<<grid, dim3(256), 0, st>>>(A, Kpad, Wpk, Kpad / 32, bias, C, ldc, nvalid, flags);
}

static void graphnorm(hipStream_t st, float* X, int F, const int* batch, const int* gs,
                      const float* w, const float* b, const float* ms,
                      float* meanb, float* varb, int act) {
  k_seg_stat<<<gblk(NG * F), 256, 0, st>>>(X, F, F, gs, meanb, 0);
  k_center<<<gblk((long long)NN * F), 256, 0, st>>>(X, F, batch, ms, meanb, NN);
  k_seg_stat<<<gblk(NG * F), 256, 0, st>>>(X, F, F, gs, varb, 1);
  k_norm_act<<<gblk((long long)NN * F), 256, 0, st>>>(X, F, batch, w, b, varb, act, NN);
}

extern "C" void kernel_launch(void* const* d_in, const int* in_sizes, int n_in,
                              void* d_out, int out_size, void* d_ws, size_t ws_size,
                              hipStream_t stream) {
  (void)in_sizes; (void)n_in; (void)out_size; (void)ws_size;

  // ---- inputs (insertion-order flattening of setup_inputs()) ----
  const float* x      = (const float*)d_in[0];
  const int*   ei     = (const int*)d_in[1];
  const int*   batch  = (const int*)d_in[2];
  const int*   e_src  = ei;
  const int*   e_dst  = ei + NE;

  int pi = 3;
  const float *sWl[5], *sBl[5], *sWr[5];
  for (int l = 0; l < 5; ++l) { sWl[l] = (const float*)d_in[pi++]; sBl[l] = (const float*)d_in[pi++]; sWr[l] = (const float*)d_in[pi++]; }
  const float *snw[4], *snb[4], *snms[4];
  for (int l = 0; l < 4; ++l) { snw[l] = (const float*)d_in[pi++]; snb[l] = (const float*)d_in[pi++]; snms[l] = (const float*)d_in[pi++]; }
  const float *gW[4], *gAs[4], *gAd[4], *gB[4];
  for (int l = 0; l < 4; ++l) { gW[l] = (const float*)d_in[pi++]; gAs[l] = (const float*)d_in[pi++]; gAd[l] = (const float*)d_in[pi++]; gB[l] = (const float*)d_in[pi++]; }
  const float *gnw[3], *gnb[3], *gnms[3];
  for (int l = 0; l < 3; ++l) { gnw[l] = (const float*)d_in[pi++]; gnb[l] = (const float*)d_in[pi++]; gnms[l] = (const float*)d_in[pi++]; }
  const float* clsW = (const float*)d_in[pi++];
  const float* clsB = (const float*)d_in[pi++];

  // ---- workspace carve-out (~190 MB) ----
  char* wsp = (char*)d_ws;
  auto carve = [&](size_t bytes) -> char* {
    char* p = wsp;
    wsp += (bytes + 255) & ~(size_t)255;
    return p;
  };
  _Float16* Abf  = (_Float16*)carve((size_t)NN * 2048 * 2);     // f16 activations (padded K)
  _Float16* Wpk  = (_Float16*)carve((size_t)2048 * 2048 * 2);   // packed weights
  float* F0      = (float*)carve((size_t)NN * 2048 * 4);
  float* F1      = (float*)carve((size_t)NN * 2048 * 4);
  float* SOUT    = (float*)carve((size_t)NN * OUTF * 4);
  float* GOUT    = (float*)carve((size_t)NN * OUTF * 4);
  float* meanb   = (float*)carve((size_t)NG * 2048 * 4);
  float* varb    = (float*)carve((size_t)NG * 2048 * 4);
  int*   cnt     = (int*)carve(NG * 4);
  int*   gs      = (int*)carve((NG + 1) * 4);
  int*   deg     = (int*)carve(NN * 4);
  float* as_     = (float*)carve((size_t)NN * HEADS_ * 4);
  float* ad_     = (float*)carve((size_t)NN * HEADS_ * 4);
  float* esc     = (float*)carve((size_t)(NE + NN) * HEADS_ * 4);
  float* den     = (float*)carve((size_t)NN * HEADS_ * 4);
  float* mx      = (float*)carve((size_t)NN * HEADS_ * 4);

  // ---- graph boundaries + in-degree (shared by all layers) ----
  k_fill_f32<<<1, 256, 0, stream>>>((float*)cnt, NG, 0.f);       // int 0 == f32 0 bits
  k_graph_count<<<gblk(NN), 256, 0, stream>>>(batch, NN, cnt);
  k_prefix<<<1, 32, 0, stream>>>(cnt, gs);
  k_fill_f32<<<gblk(NN), 256, 0, stream>>>((float*)deg, NN, 0.f);
  k_deg<<<gblk(NE), 256, 0, stream>>>(e_dst, NE, deg);

  // =========================== SAGE branch ============================
  const float* cur = x;
  int cur_ld = INF_;
  for (int li = 0; li < 5; ++li) {
    const int Kin  = (li == 0) ? INF_ : HID_;
    const int Kpad = (li == 0) ? 64 : HID_;
    const int Nout = (li == 4) ? OUTF : HID_;
    const int Npad = (li == 4) ? 128 : HID_;
    float* outb = (li == 4) ? SOUT : ((li & 1) ? F1 : F0);
    float* aggb = (li == 4) ? F0 : outb;   // agg lives in out buffer until cvt'd

    k_fill_f32<<<gblk((long long)NN * Kin), 256, 0, stream>>>(aggb, (long long)NN * Kin, 0.f);
    k_scatter_add<<<gblk((long long)NE * Kin), 256, 0, stream>>>(cur, cur_ld, e_src, e_dst, NE, Kin, aggb);
    k_div_deg<<<gblk((long long)NN * Kin), 256, 0, stream>>>(aggb, Kin, deg, NN);

    // out = agg @ Wl^T + bl
    k_cvt_f16<<<gblk((long long)NN * Kpad), 256, 0, stream>>>(aggb, Kin, Kin, Abf, Kpad, NN);
    k_pack_w<<<gblk((long long)(Npad / 16) * (Kpad / 32) * 32), 256, 0, stream>>>(sWl[li], Nout, Kin, Kpad / 32, Npad / 16, Wpk);
    run_gemm(stream, Abf, Kpad, Wpk, sBl[li], outb, Nout, Nout, Npad, /*bias*/2);
    // out += cur @ Wr^T
    k_cvt_f16<<<gblk((long long)NN * Kpad), 256, 0, stream>>>(cur, cur_ld, Kin, Abf, Kpad, NN);
    k_pack_w<<<gblk((long long)(Npad / 16) * (Kpad / 32) * 32), 256, 0, stream>>>(sWr[li], Nout, Kin, Kpad / 32, Npad / 16, Wpk);
    run_gemm(stream, Abf, Kpad, Wpk, nullptr, outb, Nout, Nout, Npad, /*acc*/1);

    if (li < 4)
      graphnorm(stream, outb, HID_, batch, gs, snw[li], snb[li], snms[li], meanb, varb, /*relu*/1);
    cur = outb; cur_ld = Nout;
  }

  // ============================ GAT branch ============================
  const float* gcur = x;
  int gld = INF_;
  for (int li = 0; li < 3; ++li) {
    const int Kin  = (li == 0) ? INF_ : 2048;
    const int Kpad = (li == 0) ? 64 : 2048;
    const int H = HEADS_, C = HID_, HC = H * C;  // 2048

    k_cvt_f16<<<gblk((long long)NN * Kpad), 256, 0, stream>>>(gcur, gld, Kin, Abf, Kpad, NN);
    k_pack_w<<<gblk((long long)(HC / 16) * (Kpad / 32) * 32), 256, 0, stream>>>(gW[li], HC, Kin, Kpad / 32, HC / 16, Wpk);
    run_gemm(stream, Abf, Kpad, Wpk, nullptr, F1, HC, HC, HC, 0);   // h -> F1

    k_attdot<<<gblk((long long)NN * H * 32), 256, 0, stream>>>(F1, HC, C, H, gAs[li], gAd[li], as_, ad_, NN);

    k_fill_f32<<<gblk((long long)NN * HC), 256, 0, stream>>>(F0, (long long)NN * HC, 0.f);
    k_fill_f32<<<gblk((long long)NN * H), 256, 0, stream>>>(mx, (long long)NN * H, -__builtin_inff());
    k_fill_f32<<<gblk((long long)NN * H), 256, 0, stream>>>(den, (long long)NN * H, 0.f);

    k_score<<<gblk(NE + NN), 256, 0, stream>>>(e_src, e_dst, NE, NN, H, as_, ad_, esc, mx);
    k_expw<<<gblk(NE + NN), 256, 0, stream>>>(e_src, e_dst, NE, NN, H, esc, mx, den);
    k_gat_agg<<<gblk((long long)(NE + NN) * HC), 256, 0, stream>>>(e_src, e_dst, NE, NN, H, C, F1, esc, den, F0);
    k_add_bias<<<gblk((long long)NN * HC), 256, 0, stream>>>(F0, HC, gB[li], NN);

    graphnorm(stream, F0, HC, batch, gs, gnw[li], gnb[li], gnms[li], meanb, varb, /*elu*/2);
    gcur = F0; gld = HC;
  }
  {   // gat4: heads=1, C=121
    const int Kpad = 2048, C = OUTF, H = 1;
    k_cvt_f16<<<gblk((long long)NN * Kpad), 256, 0, stream>>>(F0, 2048, 2048, Abf, Kpad, NN);
    k_pack_w<<<gblk((long long)(128 / 16) * (Kpad / 32) * 32), 256, 0, stream>>>(gW[3], OUTF, 2048, Kpad / 32, 128 / 16, Wpk);
    run_gemm(stream, Abf, Kpad, Wpk, nullptr, F1, OUTF, OUTF, 128, 0);   // h -> F1 (N x 121)

    k_attdot<<<gblk((long long)NN * H * 32), 256, 0, stream>>>(F1, C, C, H, gAs[3], gAd[3], as_, ad_, NN);
    k_fill_f32<<<gblk((long long)NN * OUTF), 256, 0, stream>>>(GOUT, (long long)NN * OUTF, 0.f);
    k_fill_f32<<<gblk(NN), 256, 0, stream>>>(mx, NN, -__builtin_inff());
    k_fill_f32<<<gblk(NN), 256, 0, stream>>>(den, NN, 0.f);
    k_score<<<gblk(NE + NN), 256, 0, stream>>>(e_src, e_dst, NE, NN, H, as_, ad_, esc, mx);
    k_expw<<<gblk(NE + NN), 256, 0, stream>>>(e_src, e_dst, NE, NN, H, esc, mx, den);
    k_gat_agg<<<gblk((long long)(NE + NN) * C), 256, 0, stream>>>(e_src, e_dst, NE, NN, H, C, F1, esc, den, GOUT);
    k_add_bias<<<gblk((long long)NN * OUTF), 256, 0, stream>>>(GOUT, OUTF, gB[3], NN);
  }

  // ========================== classifier =============================
  k_concat_cvt<<<gblk((long long)NN * 256), 256, 0, stream>>>(SOUT, GOUT, Abf);
  k_pack_w<<<gblk((long long)(128 / 16) * (256 / 32) * 32), 256, 0, stream>>>(clsW, OUTF, 2 * OUTF, 256 / 32, 128 / 16, Wpk);
  run_gemm(stream, Abf, 256, Wpk, clsB, (float*)d_out, OUTF, OUTF, 128, /*bias*/2);
}
